// EarthAttention3D_3633542332427
// MI455X (gfx1250) — matrix operand order repaired and checked by reading, not compiled, not run on hardware
//
#include <hip/hip_runtime.h>
#include <hip/hip_bf16.h>

typedef __attribute__((ext_vector_type(16))) _Float16 v16h;
typedef __attribute__((ext_vector_type(8)))  _Float16 v8h;
typedef __attribute__((ext_vector_type(4)))  _Float16 v4h;
typedef __attribute__((ext_vector_type(8)))  float    v8f;

#define WZ 2
#define WH 6
#define WW 12
#define N_TOK 144            // WZ*WH*WW
#define DMODEL 192
#define HEADS 6
#define HD 32
#define NW 64
#define NPAD 160             // padded key length for P@V (5 x K=32)
#define SCALE 0.17677669529663687f   // 1/sqrt(32)

// LDS layout (in _Float16 elements)
#define OFF_X   0                               // [144][192] x, later attn_out
#define OFF_Q   (OFF_X + N_TOK * DMODEL)        // [6][144][32]  (pre-scaled)
#define OFF_K   (OFF_Q + HEADS * N_TOK * HD)    // [6][144][32]
#define OFF_VT  (OFF_K + HEADS * N_TOK * HD)    // [6][32][160]  V transposed, k-padded
#define OFF_S   (OFF_VT + HEADS * HD * NPAD)    // per-wave [16][160] scores/probs
#define SMEM_HALFS (OFF_S + 8 * 16 * NPAD)
#define SMEM_BYTES (SMEM_HALFS * 2)             // 268,288 B < 320 KB

// ---- A-matrix fragment (16x32 f16): contiguous layout matching frag_B:
//      lane half kb = (lane>=16)?16:0 ; values K = {kb+0..kb+15}
__device__ __forceinline__ v16h frag_A(const _Float16* rowptr, int lane) {
  const int kb = (lane >= 16) ? 16 : 0;
  v16h a;
#pragma unroll
  for (int j = 0; j < 16; ++j) {
    a[j]     = rowptr[kb + j];
  }
  return a;
}

// ---- B-matrix fragment (32x16 f16): contiguous k order matching frag_A; each
//      half-wave holds 16 *consecutive* K values: K = ((lane>=16)?16:0) + j.
//      rowptr points at the (column n = lane&15) data stored K-contiguous.
__device__ __forceinline__ v16h frag_B(const _Float16* rowptr, int lane) {
  const int kb = (lane >= 16) ? 16 : 0;
  v16h b;
#pragma unroll
  for (int j = 0; j < 16; ++j) b[j] = rowptr[kb + j];
  return b;
}

// B fragment gathered from a row-major f32 weight matrix (column `colptr`, ld = row stride)
__device__ __forceinline__ v16h frag_B_cols_f32(const float* colptr, int ld, int lane) {
  const int kb = (lane >= 16) ? 16 : 0;
  v16h b;
#pragma unroll
  for (int j = 0; j < 16; ++j) b[j] = (_Float16)colptr[(kb + j) * ld];
  return b;
}

// Wave-local LDS producer->consumer fence (CDNA5 split DS counter).
__device__ __forceinline__ void wave_lds_fence() {
  asm volatile("s_wait_dscnt 0" ::: "memory");
}

__global__ void __launch_bounds__(256, 1)
earth_attn3d_fused(const float* __restrict__ x, const float* __restrict__ mask,
                   const float* __restrict__ W1, const float* __restrict__ b1,
                   const float* __restrict__ W2, const float* __restrict__ b2,
                   const float* __restrict__ bias_table, const int* __restrict__ rollp,
                   float* __restrict__ out) {
  extern __shared__ _Float16 smem[];
  _Float16* sX  = smem + OFF_X;
  _Float16* sQ  = smem + OFF_Q;
  _Float16* sK  = smem + OFF_K;
  _Float16* sVt = smem + OFF_VT;

  const int b    = blockIdx.x;
  const int tid  = threadIdx.x;
  // readfirstlane => compiler knows these are wave-uniform (scalar branches, not exec masks)
  const int wave = __builtin_amdgcn_readfirstlane(tid >> 5);
  const int lane = tid & 31;
  const int nl   = lane & 15;
  const int hi   = (lane >= 16) ? 1 : 0;
  const int roff = hi * 8;
  const int roll = rollp[0];

  _Float16* sS = smem + OFF_S + wave * (16 * NPAD);

  // ---------------- Phase 0: stage x -> LDS f16 (vectorized); zero V k-padding ----
  {
    const float4* xb4 = (const float4*)(x + (size_t)b * (N_TOK * DMODEL));
    v4h* sX4 = (v4h*)sX;
    for (int i = tid; i < (N_TOK * DMODEL) / 4; i += 256) {
      const float4 f = xb4[i];
      v4h h;
      h[0] = (_Float16)f.x; h[1] = (_Float16)f.y;
      h[2] = (_Float16)f.z; h[3] = (_Float16)f.w;
      sX4[i] = h;
    }
    // zero sVt[:, :, 144..159]  (384 eight-half chunks)
    const v8h z = {};
    for (int i = tid; i < HEADS * HD * 2; i += 256) {
      const int hd8 = i >> 1, half = i & 1;
      *(v8h*)(sVt + hd8 * NPAD + N_TOK + half * 8) = z;
    }
  }
  __syncthreads();

  // ---------------- Phase 1: qkv = x @ W1 + b1  (36 col-tiles over 8 waves) -------
  for (int nt = wave; nt < 36; nt += 8) {
    const int ncol0 = nt * 16;         // scalar
    const int col   = ncol0 + nl;
    v16h bf[6];
#pragma unroll
    for (int kc = 0; kc < 6; ++kc)
      bf[kc] = frag_B_cols_f32(W1 + (size_t)(kc * 32) * (3 * DMODEL) + col, 3 * DMODEL, lane);
    const float bias1 = b1[col];
    for (int mt = 0; mt < 9; ++mt) {
      v16h af[6];
#pragma unroll
      for (int kc = 0; kc < 6; ++kc)
        af[kc] = frag_A(sX + (mt * 16 + nl) * DMODEL + kc * 32, lane);
      v8f acc = {};
#pragma unroll
      for (int kc = 0; kc < 6; ++kc)
        acc = __builtin_amdgcn_wmma_f32_16x16x32_f16(false, af[kc], false, bf[kc],
                                                     (short)0, acc, false, false);
      // scalar 3-way routing: whole 16-col tile belongs to exactly one of q/k/v
      if (ncol0 < DMODEL) {            // Q (pre-scaled)
        const int h = col >> 5, d = col & 31;
#pragma unroll
        for (int r = 0; r < 8; ++r) {
          const int n = mt * 16 + r + roff;
          sQ[(h * N_TOK + n) * HD + d] = (_Float16)((acc[r] + bias1) * SCALE);
        }
      } else if (ncol0 < 2 * DMODEL) { // K (row-major: K^T columns are contiguous)
        const int c = col - DMODEL, h = c >> 5, d = c & 31;
#pragma unroll
        for (int r = 0; r < 8; ++r) {
          const int n = mt * 16 + r + roff;
          sK[(h * N_TOK + n) * HD + d] = (_Float16)(acc[r] + bias1);
        }
      } else {                         // V stored transposed [h][d][k] (b128 store over r)
        const int c = col - 2 * DMODEL, h = c >> 5, d = c & 31;
#pragma unroll
        for (int r = 0; r < 8; ++r) {
          const int n = mt * 16 + r + roff;
          sVt[(h * HD + d) * NPAD + n] = (_Float16)(acc[r] + bias1);
        }
      }
    }
  }
  __syncthreads();

  // ---------------- Phase 2+3: attention, 54 (head, row-tile) tasks over 8 waves ---
  const float* maskb = mask + (size_t)(b & (NW - 1)) * (N_TOK * N_TOK);
  for (int task = wave; task < HEADS * 9; task += 8) {
    const int h  = task / 9;           // scalar
    const int mt = task % 9;           // scalar

    // S = q @ k^T + rel-pos bias (+ window mask); q fragment invariant over nt2
    const v16h qa = frag_A(sQ + (h * N_TOK + mt * 16 + nl) * HD, lane);
    for (int nt2 = 0; nt2 < 9; ++nt2) {
      const v16h kb = frag_B(sK + (h * N_TOK + nt2 * 16 + nl) * HD, lane);
      v8f s = {};
      s = __builtin_amdgcn_wmma_f32_16x16x32_f16(false, qa, false, kb, (short)0, s, false, false);
      const int j  = nt2 * 16 + nl;
      const int jz = j / (WH * WW), jh = (j / WW) % WH, jw = j % WW;
#pragma unroll
      for (int r = 0; r < 8; ++r) {
        const int i  = mt * 16 + r + roff;
        const int iz = i / (WH * WW), ih = (i / WW) % WH, iw = i % WW;
        const int bidx = (iz + WZ * jz) * ((2 * WW - 1) * WH * WH)
                       + (ih + WH * jh) * (2 * WW - 1)
                       + (iw - jw + WW - 1);
        float sv = s[r] + bias_table[bidx * HEADS + h];
        if (roll) sv += maskb[i * N_TOK + j];
        sS[(r + roff) * NPAD + j] = (_Float16)sv;
      }
    }
    wave_lds_fence();

    // softmax: row nl handled by lane pair (nl, nl+16); each half-wave owns
    // 72 columns (9 x v8h); max/sum combined with one cross-half shuffle.
    {
      _Float16* row  = sS + nl * NPAD;
      _Float16* half = row + hi * 72;   // lanes 0..15: cols 0..71 ; 16..31: 72..143
      v8h c[9];
      float m = -1e30f;
#pragma unroll
      for (int t = 0; t < 9; ++t) {
        c[t] = *(const v8h*)(half + t * 8);
#pragma unroll
        for (int j = 0; j < 8; ++j) m = fmaxf(m, (float)c[t][j]);
      }
      m = fmaxf(m, __shfl_xor(m, 16, 32));        // row max across the lane pair
      float sum = 0.f;
#pragma unroll
      for (int t = 0; t < 9; ++t) {
#pragma unroll
        for (int j = 0; j < 8; ++j) {
          const float e = __expf((float)c[t][j] - m);
          sum += e;
          c[t][j] = (_Float16)e;
        }
      }
      sum += __shfl_xor(sum, 16, 32);             // row sum across the lane pair
      const float inv = 1.f / sum;
#pragma unroll
      for (int t = 0; t < 9; ++t) {
#pragma unroll
        for (int j = 0; j < 8; ++j) c[t][j] = (_Float16)((float)c[t][j] * inv);
        *(v8h*)(half + t * 8) = c[t];
      }
      if (hi) {                                   // zero the k-padding cols 144..159
        const v8h z = {};
        *(v8h*)(row + N_TOK) = z;
        *(v8h*)(row + N_TOK + 8) = z;
      }
    }
    wave_lds_fence();

    // O = P @ V   (K=160 = 5 x 32, padded tail contributes 0)
    v16h pa[5];
#pragma unroll
    for (int kc = 0; kc < 5; ++kc)
      pa[kc] = frag_A(sS + nl * NPAD + kc * 32, lane);
#pragma unroll
    for (int dt = 0; dt < 2; ++dt) {
      v8f o = {};
#pragma unroll
      for (int kc = 0; kc < 5; ++kc) {
        const v16h vb = frag_B(sVt + (h * HD + dt * 16 + nl) * NPAD + kc * 32, lane);
        o = __builtin_amdgcn_wmma_f32_16x16x32_f16(false, pa[kc], false, vb,
                                                   (short)0, o, false, false);
      }
#pragma unroll
      for (int r = 0; r < 8; ++r) {
        const int n = mt * 16 + r + roff;
        sX[n * DMODEL + h * HD + dt * 16 + nl] = (_Float16)o[r];   // attn_out reuses sX
      }
    }
  }
  __syncthreads();

  // ---------------- Phase 4: out = attn_out @ W2 + b2 (fp32 to HBM) ---------------
  float* outb = out + (size_t)b * (N_TOK * DMODEL);
  for (int ct = wave; ct < 12; ct += 8) {
    const int col = ct * 16 + nl;
    v16h bf[6];
#pragma unroll
    for (int kc = 0; kc < 6; ++kc)
      bf[kc] = frag_B_cols_f32(W2 + (size_t)(kc * 32) * DMODEL + col, DMODEL, lane);
    const float bias2 = b2[col];
    for (int mt = 0; mt < 9; ++mt) {
      v16h af[6];
#pragma unroll
      for (int kc = 0; kc < 6; ++kc)
        af[kc] = frag_A(sX + (mt * 16 + nl) * DMODEL + kc * 32, lane);
      v8f acc = {};
#pragma unroll
      for (int kc = 0; kc < 6; ++kc)
        acc = __builtin_amdgcn_wmma_f32_16x16x32_f16(false, af[kc], false, bf[kc],
                                                     (short)0, acc, false, false);
#pragma unroll
      for (int r = 0; r < 8; ++r) {
        const int n = mt * 16 + r + roff;
        outb[n * DMODEL + col] = acc[r] + bias2;
      }
    }
  }
}

extern "C" void kernel_launch(void* const* d_in, const int* in_sizes, int n_in,
                              void* d_out, int out_size, void* d_ws, size_t ws_size,
                              hipStream_t stream) {
  (void)in_sizes; (void)n_in; (void)d_ws; (void)ws_size; (void)out_size;
  const float* x    = (const float*)d_in[0];
  const float* mask = (const float*)d_in[1];
  const float* W1   = (const float*)d_in[2];
  const float* b1   = (const float*)d_in[3];
  const float* W2   = (const float*)d_in[4];
  const float* b2   = (const float*)d_in[5];
  const float* bt   = (const float*)d_in[6];
  const int*   roll = (const int*)d_in[7];

  (void)hipFuncSetAttribute(reinterpret_cast<const void*>(earth_attn3d_fused),
                            hipFuncAttributeMaxDynamicSharedMemorySize, SMEM_BYTES);

  earth_attn3d_fused<<<1024, 256, SMEM_BYTES, stream>>>(
      x, mask, W1, b1, W2, b2, bt, roll, (float*)d_out);
}
